// TemporalDecoder_18399639896602
// MI455X (gfx1250) — matrix operand order repaired and checked by reading, not compiled, not run on hardware
//
#include <hip/hip_runtime.h>
#include <hip/hip_bf16.h>

// GRU decoder for MI455X (gfx1250), wave32 + WMMA.
// Kernel 1: serial recurrence on ONE WGP, 16 waves x 6 gh-tiles.
//   - 4/6 f16 B-tiles per wave pinned in VGPRs (fits 256-VGPR cap, no spills),
//     2/6 in LDS in WMMA layout (compiler software-pipelines their ds loads).
//   - h state: f32 master copy in LDS (thread-private access) + f16 copy in
//     WMMA-A layout (hA0/hA1), so the per-chunk A operand is 2x ds_load_b128
//     with zero per-step conversions on the serial critical path.
//   - per-step 64x384x128 GEMM via v_wmma_f32_16x16x32_f16, fused gate math,
//     history streamed to d_ws as f16 (stays in 192 MB L2 for kernel 2).
// Kernel 2: bandwidth-bound output GEMM (hs f16 @ fc_W f16 -> f32), WMMA,
//   non-temporal stores for the 268 MB output stream.

#define SEQ   2048
#define HID   128
#define BAT   64
#define NOUT  512
#define GH_STRIDE 388   // padded so M and M+8 rows of a C-tile hit disjoint LDS banks
#define HB_STRIDE 132   // padded h-state stride
#define NLDSTILE 8      // 8 B-tiles (2 per N-group) kept in LDS

typedef __attribute__((ext_vector_type(16))) _Float16 v16h;
typedef __attribute__((ext_vector_type(8)))  _Float16 v8h;
typedef __attribute__((ext_vector_type(8)))  float    v8f;

__device__ __forceinline__ float gru_elem(float gr, float gz, float gn, float h,
                                          float br, float bz, float bi, float bh) {
    float r = 1.0f / (1.0f + __expf(-(br + gr)));
    float z = 1.0f / (1.0f + __expf(-(bz + gz)));
    float e = __expf(2.0f * (bi + r * (gn + bh)));
    float n = 1.0f - 2.0f / (e + 1.0f);            // tanh, saturates correctly at +/-inf
    return (1.0f - z) * n + z * h;
}

// Build a WMMA f16 B-operand lane-slice from a W_hh row (f32 at kb+0..7 and kb+16..23).
// B(k,n) = Whh[n][k]; lanes 0-15 hold col n (K=kc*32+0..7,16..23), lanes 16-31 K+8.
__device__ __forceinline__ v16h load_b_slice(const float* __restrict__ Whh,
                                             int nrow, int kb) {
    const float4* wp = (const float4*)(Whh + nrow * HID + kb);
    v16h bb;
    #pragma unroll
    for (int q = 0; q < 4; ++q) {
        float4 v = wp[q + (q & 2)];
        bb[4*q + 0] = (_Float16)v.x;  bb[4*q + 1] = (_Float16)v.y;
        bb[4*q + 2] = (_Float16)v.z;  bb[4*q + 3] = (_Float16)v.w;
    }
    return bb;
}

// ---------------------------------------------------------------------------
// Kernel 1: recurrence. 1 block x 512 threads (16 waves).
// Wave w: M-block mb = w&3, N-tiles nb = (w>>2)*6 .. +5.
// Tiles 0..3 resident in VGPRs; tiles 4..5 streamed from LDS each step.
// ---------------------------------------------------------------------------
__global__ __launch_bounds__(512, 1)
void gru_recurrence(const float* __restrict__ latent,
                    const float* __restrict__ Whh,
                    const float* __restrict__ b_ih,
                    const float* __restrict__ b_hh,
                    _Float16*    __restrict__ hs)
{
    extern __shared__ float sm[];
    float* gh   = sm;                         // [64][GH_STRIDE] f32 gate pre-activations
    float* hbuf = sm + BAT * GH_STRIDE;       // [64][HB_STRIDE] f32 hidden state (master)
    float* cbr  = hbuf + BAT * HB_STRIDE;     // folded biases
    float* cbz  = cbr + HID;
    float* binb = cbz + HID;
    float* bhnb = binb + HID;
    // LDS-resident B tiles, WMMA layout, split lo/hi v8h (16-B slots, aligned)
    v8h* bls0 = (v8h*)(bhnb + HID);           // [NLDSTILE*4][32]
    v8h* bls1 = bls0 + NLDSTILE * 4 * 32;
    // f16 h state in WMMA-A layout: hA0 = halves j0..7, hA1 = j8..15
    v8h* hA0  = bls1 + NLDSTILE * 4 * 32;     // [4*4][32]  (mb, kc, lane)
    v8h* hA1  = hA0 + 4 * 4 * 32;

    const int tid  = threadIdx.x;
    const int lane = tid & 31;
    const int wave = tid >> 5;

    const int mb  = wave & 3;
    const int nb0 = (wave >> 2) * 6;
    const int grp = wave >> 2;

    // elementwise-phase mapping (loop-invariant)
    const int ew_b  = tid >> 3;          // batch row
    const int ew_i0 = (tid & 7) * 16;    // 16 consecutive hidden channels
    const int mbb   = ew_b >> 4;
    const int kcc   = ew_i0 >> 5;
    v8h* hAx        = ((ew_i0 >> 4) & 1) ? hA1 : hA0;    // which half of the K-chunk
    const int slotL = (mbb * 4 + kcc) * 32 + (ew_b & 15);
    const int slotH = slotL + 16;
    _Float16* hrow  = hs + (size_t)ew_b * SEQ * HID + ew_i0;

    // ---- setup: fold biases (r/z fold b_ih+b_hh; n gate keeps them split) ----
    if (tid < HID) {
        cbr[tid]  = b_ih[tid]         + b_hh[tid];
        cbz[tid]  = b_ih[HID + tid]   + b_hh[HID + tid];
        binb[tid] = b_ih[2*HID + tid];
        bhnb[tid] = b_hh[2*HID + tid];
    }
    // ---- setup: h0 into LDS (f32 master + f16 A-layout copy) ----
    {
        const float4* src = (const float4*)(latent + ew_b * HID + ew_i0);
        float4*       dst = (float4*)(hbuf + ew_b * HB_STRIDE + ew_i0);
        v8h hlo, hhi;
        #pragma unroll
        for (int q = 0; q < 4; ++q) {
            float4 v = src[q];
            dst[q] = v;
            v8h& d = (q < 2) ? hlo : hhi;
            int  o = (q & 1) * 4;
            d[o+0] = (_Float16)v.x; d[o+1] = (_Float16)v.y;
            d[o+2] = (_Float16)v.z; d[o+3] = (_Float16)v.w;
        }
        hAx[slotL] = hlo;
        hAx[slotH] = hhi;
    }

    // ---- setup: resident B tiles (0..3) -> VGPRs for all 2048 steps ----
    v16h Breg[4][4];
    #pragma unroll
    for (int t4 = 0; t4 < 4; ++t4) {
        int nrow = (nb0 + t4) * 16 + (lane & 15);
        #pragma unroll
        for (int kc = 0; kc < 4; ++kc)
            Breg[t4][kc] = load_b_slice(Whh, nrow, kc * 32 + ((lane >> 4) << 3));
    }
    // ---- setup: LDS B tiles (4..5), written once by the mb==0 wave of each group ----
    if (mb == 0) {
        #pragma unroll
        for (int tt = 0; tt < 2; ++tt) {
            int nrow = (nb0 + 4 + tt) * 16 + (lane & 15);
            int ti   = grp * 2 + tt;
            #pragma unroll
            for (int kc = 0; kc < 4; ++kc) {
                v16h bb = load_b_slice(Whh, nrow, kc * 32 + ((lane >> 4) << 3));
                int slot = (ti * 4 + kc) * 32 + lane;
                bls0[slot] = __builtin_shufflevector(bb, bb, 0,1,2,3,4,5,6,7);
                bls1[slot] = __builtin_shufflevector(bb, bb, 8,9,10,11,12,13,14,15);
            }
        }
    }
    __syncthreads();

    for (int t = 0; t < SEQ; ++t) {
        // ---------- WMMA phase: gh = h @ Whh^T (no bias) ----------
        v8f acc[6];
        #pragma unroll
        for (int t6 = 0; t6 < 6; ++t6)
            acc[t6] = (v8f){0.f,0.f,0.f,0.f,0.f,0.f,0.f,0.f};

        #pragma unroll
        for (int kc = 0; kc < 4; ++kc) {
            // A operand: 2x ds_load_b128 from the f16 A-layout buffer, no cvt
            int aslot = (mb * 4 + kc) * 32 + lane;
            v8h alo = hA0[aslot];
            v8h ahi = hA1[aslot];
            // staged LDS B tiles for this chunk
            int s4 = ((grp * 2 + 0) * 4 + kc) * 32 + lane;
            int s5 = ((grp * 2 + 1) * 4 + kc) * 32 + lane;
            v8h b4lo = bls0[s4], b4hi = bls1[s4];
            v8h b5lo = bls0[s5], b5hi = bls1[s5];

            v16h a = __builtin_shufflevector(alo, ahi, 0,1,2,3,4,5,6,7,8,9,10,11,12,13,14,15);
            #pragma unroll
            for (int t4 = 0; t4 < 4; ++t4)
                acc[t4] = __builtin_amdgcn_wmma_f32_16x16x32_f16(
                    false, a, false, Breg[t4][kc], (short)0, acc[t4], false, false);

            v16h b4 = __builtin_shufflevector(b4lo, b4hi, 0,1,2,3,4,5,6,7,8,9,10,11,12,13,14,15);
            v16h b5 = __builtin_shufflevector(b5lo, b5hi, 0,1,2,3,4,5,6,7,8,9,10,11,12,13,14,15);
            acc[4] = __builtin_amdgcn_wmma_f32_16x16x32_f16(
                false, a, false, b4, (short)0, acc[4], false, false);
            acc[5] = __builtin_amdgcn_wmma_f32_16x16x32_f16(
                false, a, false, b5, (short)0, acc[5], false, false);
        }
        // C layout: element r of lane l is row M = r + 8*(l>>4), col N = l&15
        {
            int rb = mb * 16 + ((lane >> 4) << 3);
            #pragma unroll
            for (int t6 = 0; t6 < 6; ++t6) {
                int col = (nb0 + t6) * 16 + (lane & 15);
                #pragma unroll
                for (int r = 0; r < 8; ++r)
                    gh[(rb + r) * GH_STRIDE + col] = acc[t6][r];
            }
        }
        __syncthreads();

        // ---------- fused gate math + state update ----------
        {
            const float* grp_p = gh   + ew_b * GH_STRIDE + ew_i0;
            float*       hbp   = hbuf + ew_b * HB_STRIDE + ew_i0;
            v8h hlo = {}, hhi = {};
            #pragma unroll
            for (int q = 0; q < 4; ++q) {
                float4 g_r = *(const float4*)(grp_p + 4*q);
                float4 g_z = *(const float4*)(grp_p + HID   + 4*q);
                float4 g_n = *(const float4*)(grp_p + 2*HID + 4*q);
                float4 h_o = *(const float4*)(hbp + 4*q);
                float4 c_r = *(const float4*)(cbr  + ew_i0 + 4*q);
                float4 c_z = *(const float4*)(cbz  + ew_i0 + 4*q);
                float4 c_i = *(const float4*)(binb + ew_i0 + 4*q);
                float4 c_h = *(const float4*)(bhnb + ew_i0 + 4*q);
                float4 hn4;
                hn4.x = gru_elem(g_r.x, g_z.x, g_n.x, h_o.x, c_r.x, c_z.x, c_i.x, c_h.x);
                hn4.y = gru_elem(g_r.y, g_z.y, g_n.y, h_o.y, c_r.y, c_z.y, c_i.y, c_h.y);
                hn4.z = gru_elem(g_r.z, g_z.z, g_n.z, h_o.z, c_r.z, c_z.z, c_i.z, c_h.z);
                hn4.w = gru_elem(g_r.w, g_z.w, g_n.w, h_o.w, c_r.w, c_z.w, c_i.w, c_h.w);
                *(float4*)(hbp + 4*q) = hn4;
                v8h& d = (q < 2) ? hlo : hhi;
                int  o = (q & 1) * 4;
                d[o+0] = (_Float16)hn4.x; d[o+1] = (_Float16)hn4.y;
                d[o+2] = (_Float16)hn4.z; d[o+3] = (_Float16)hn4.w;
            }
            // f16 h -> WMMA-A layout buffer for next step's GEMM
            hAx[slotL] = hlo;
            hAx[slotH] = hhi;
            // history -> global (f16, row-major [b][t][k]); stays in L2 for kernel 2
            v8h* gdst = (v8h*)(hrow + (size_t)t * HID);
            gdst[0] = hlo;
            gdst[1] = hhi;
        }
        __syncthreads();
    }
}

// ---------------------------------------------------------------------------
// Kernel 2: out = hs @ fc_W^T + fc_b.  M = 64*2048, N = 512, K = 128.
// One block (8 waves) per 16-row M-tile; wave w covers N-tiles 4w..4w+3.
// ---------------------------------------------------------------------------
__global__ __launch_bounds__(256)
void out_gemm(const _Float16* __restrict__ hsf,
              const float*    __restrict__ fcW,
              const float*    __restrict__ fcb,
              float*          __restrict__ outp)
{
    const int lane  = threadIdx.x & 31;
    const int wave  = threadIdx.x >> 5;
    const int mtile = blockIdx.x;            // 0..8191
    const int ngrp  = wave;                  // 0..7 -> 4 N-tiles each

    // A operand (f16 from global, row-major): contiguous K order k = 16*(lane>>4) + s, same as B below (not kernel 1's)
    v16h A[4];
    {
        const _Float16* arow = hsf + (size_t)(mtile * 16 + (lane & 15)) * HID;
        #pragma unroll
        for (int kc = 0; kc < 4; ++kc) {
            const v8h* p = (const v8h*)(arow + kc*32 + ((lane >> 4) << 4));
            v8h a0 = p[0];          // K offsets +0..7
            v8h a1 = p[1];          // K offsets +8..15
            A[kc] = __builtin_shufflevector(a0, a1, 0,1,2,3,4,5,6,7,8,9,10,11,12,13,14,15);
        }
    }

    #pragma unroll
    for (int u = 0; u < 4; ++u) {
        int nt   = ngrp * 4 + u;
        int ncol = nt * 16 + (lane & 15);
        v8f acc = (v8f){0.f,0.f,0.f,0.f,0.f,0.f,0.f,0.f};
        #pragma unroll
        for (int kc = 0; kc < 4; ++kc) {
            const float4* wp = (const float4*)(fcW + ncol * HID + kc*32 + ((lane >> 4) << 4));
            v16h bb;
            #pragma unroll
            for (int q = 0; q < 4; ++q) {
                float4 v = wp[q];
                bb[4*q + 0] = (_Float16)v.x;  bb[4*q + 1] = (_Float16)v.y;
                bb[4*q + 2] = (_Float16)v.z;  bb[4*q + 3] = (_Float16)v.w;
            }
            acc = __builtin_amdgcn_wmma_f32_16x16x32_f16(false, A[kc], false, bb,
                                                         (short)0, acc, false, false);
        }
        float bv = fcb[ncol];
        int rbase = mtile * 16 + ((lane >> 4) << 3);
        #pragma unroll
        for (int r = 0; r < 8; ++r)
            __builtin_nontemporal_store(acc[r] + bv,
                                        outp + (size_t)(rbase + r) * NOUT + ncol);
    }
}

extern "C" void kernel_launch(void* const* d_in, const int* in_sizes, int n_in,
                              void* d_out, int out_size, void* d_ws, size_t ws_size,
                              hipStream_t stream) {
    const float* latent = (const float*)d_in[0];
    const float* Whh    = (const float*)d_in[1];
    const float* b_ih   = (const float*)d_in[2];
    const float* b_hh   = (const float*)d_in[3];
    const float* fcW    = (const float*)d_in[4];
    const float* fcb    = (const float*)d_in[5];
    float*       outp   = (float*)d_out;
    _Float16*    hs     = (_Float16*)d_ws;   // needs 64*2048*128*2 B = 32 MB

    const size_t lds_bytes = (size_t)(BAT * GH_STRIDE + BAT * HB_STRIDE + 4 * HID) * 4
                           + (size_t)NLDSTILE * 4 * 32 * 2 * 16   // B tiles lo+hi
                           + (size_t)4 * 4 * 32 * 2 * 16;         // hA0 + hA1
    (void)hipFuncSetAttribute((const void*)gru_recurrence,
                              hipFuncAttributeMaxDynamicSharedMemorySize,
                              (int)lds_bytes);

    gru_recurrence<<<1, 512, lds_bytes, stream>>>(latent, Whh, b_ih, b_hh, hs);
    out_gemm<<<(BAT * SEQ) / 16, 256, 0, stream>>>(hs, fcW, fcb, outp);
    (void)in_sizes; (void)n_in; (void)out_size; (void)ws_size;
}
